// TransformerGNN_67920612819555
// MI455X (gfx1250) — matrix operand order repaired and checked by reading, not compiled, not run on hardware
//
#include <hip/hip_runtime.h>
#include <hip/hip_bf16.h>
#include <cstdint>
#include <cstddef>

// ---------------------------------------------------------------------------
// PointTransformer GNN pipeline for MI455X (gfx1250, wave32, WMMA).
// Dense GEMMs: v_wmma_f32_16x16x32_f16, 32x(16*NT) register-blocked per wave.
// kNN staging: global_load_async_to_lds_b32 + s_wait_asynccnt (ASYNCcnt path).
// ---------------------------------------------------------------------------

typedef _Float16 h16;
typedef _Float16 v16h __attribute__((ext_vector_type(16)));
typedef _Float16 v8h  __attribute__((ext_vector_type(8)));
typedef float    v8f  __attribute__((ext_vector_type(8)));

#define KNBR 16

// ---------------- kNN: one lane per query point, async-LDS-tiled candidates
__global__ void knn_kernel(const float* __restrict__ pos, int N,
                           int* __restrict__ idx) {
    __shared__ float sx[256], sy[256], sz[256];
    const int i = blockIdx.x * blockDim.x + threadIdx.x;
    float px = 0.f, py = 0.f, pz = 0.f;
    if (i < N) { px = pos[3*i]; py = pos[3*i+1]; pz = pos[3*i+2]; }
    float bd[KNBR]; int bi[KNBR];
#pragma unroll
    for (int t = 0; t < KNBR; ++t) { bd[t] = 3.0e38f; bi[t] = 0; }

    for (int tile = 0; tile < N; tile += 256) {
        const int j = tile + threadIdx.x;
        if (j < N) {
            // CDNA5 async copy global -> LDS (ASYNCcnt), per-lane addresses.
            const unsigned lx = (unsigned)(uintptr_t)&sx[threadIdx.x];
            const unsigned ly = (unsigned)(uintptr_t)&sy[threadIdx.x];
            const unsigned lz = (unsigned)(uintptr_t)&sz[threadIdx.x];
            const unsigned long long ga =
                (unsigned long long)(uintptr_t)(pos + 3*j);
            asm volatile("global_load_async_to_lds_b32 %0, %1, off"
                         :: "v"(lx), "v"(ga) : "memory");
            asm volatile("global_load_async_to_lds_b32 %0, %1, off offset:4"
                         :: "v"(ly), "v"(ga) : "memory");
            asm volatile("global_load_async_to_lds_b32 %0, %1, off offset:8"
                         :: "v"(lz), "v"(ga) : "memory");
        }
        asm volatile("s_wait_asynccnt 0x0" ::: "memory");
        __syncthreads();
        const int lim = min(256, N - tile);
        for (int t = 0; t < lim; ++t) {
            const int j2 = tile + t;
            const float dx = px - sx[t], dy = py - sy[t], dz = pz - sz[t];
            const float d = dx*dx + dy*dy + dz*dz;
            if (j2 != i && d < bd[KNBR-1]) {           // worst-of-16 early out
                float cd = d; int cj = j2;             // bubble insert (regs)
#pragma unroll
                for (int p = 0; p < KNBR; ++p) {
                    if (cd < bd[p]) {
                        float td = bd[p]; int ti = bi[p];
                        bd[p] = cd; bi[p] = cj; cd = td; cj = ti;
                    }
                }
            }
        }
        __syncthreads();
    }
    if (i < N) {
#pragma unroll
        for (int t = 0; t < KNBR; ++t) idx[i*KNBR + t] = bi[t];
    }
}

// ---------------- WMMA GEMM: Out[M,N] = act(A[M,K] * Bt[K,N] + bias) --------
// A: f16 row-major, lda mult of 32.  Bt: f16 [K,N] (pre-transposed weight).
// One wave computes a 32 x (16*NT) output block: MT=2 row tiles reuse each B
// fragment, NT column tiles reuse each A fragment -> 2*NT wmma per k-step.
template<int ACT, bool OUT16, int NT>
__global__ void gemm_wmma(const h16* __restrict__ A, int lda,
                          const h16* __restrict__ Bt,
                          const float* __restrict__ bias,
                          void* __restrict__ Out, int ldo,
                          int M, int N, int K) {
    const int lane  = threadIdx.x & 31;
    const int wave  = blockIdx.x * (blockDim.x >> 5) + (threadIdx.x >> 5);
    const int ntg   = N / (16 * NT);
    const int tiles = (M >> 5) * ntg;
    if (wave >= tiles) return;
    const int tm0 = (wave / ntg) * 2;
    const int tn0 = (wave % ntg) * NT;
    const int lh = lane & 15, hi = lane >> 4;

    v8f acc[2][NT] = {};
    // A fragment: lane row = tm*16+lh; hi-half lanes hold K+8..15 / K+24..31
    const h16* Ap0 = A + (size_t)(tm0*16 + lh) * lda + hi*8;
    const h16* Ap1 = Ap0 + (size_t)16 * lda;
    // B fragment: lane l holds column col+lh, rows kb+hi*8+0..7 and kb+16+hi*8+0..7
    const h16* Bp  = Bt + (size_t)(hi*8) * N + tn0*16 + lh;

    for (int kb = 0; kb < K; kb += 32) {
        __builtin_prefetch((const void*)(Ap0 + kb + 512), 0, 1);
        v16h af0, af1;
        {
            v8h a0 = *(const v8h*)(Ap0 + kb);
            v8h a1 = *(const v8h*)(Ap0 + kb + 16);
            v8h a2 = *(const v8h*)(Ap1 + kb);
            v8h a3 = *(const v8h*)(Ap1 + kb + 16);
#pragma unroll
            for (int t = 0; t < 8; ++t) {
                af0[t] = a0[t]; af0[t+8] = a1[t];
                af1[t] = a2[t]; af1[t+8] = a3[t];
            }
        }
        const h16* bp = Bp + (size_t)kb * N;
#pragma unroll
        for (int j = 0; j < NT; ++j) {
            v16h bf;
#pragma unroll
            for (int t = 0; t < 8; ++t) { bf[t] = bp[(size_t)t*N + j*16]; bf[t+8] = bp[(size_t)(16+t)*N + j*16]; }
            acc[0][j] = __builtin_amdgcn_wmma_f32_16x16x32_f16(
                            false, af0, false, bf, (short)0, acc[0][j], false, false);
            acc[1][j] = __builtin_amdgcn_wmma_f32_16x16x32_f16(
                            false, af1, false, bf, (short)0, acc[1][j], false, false);
        }
    }
#pragma unroll
    for (int i = 0; i < 2; ++i) {
#pragma unroll
        for (int j = 0; j < NT; ++j) {
            const int col = (tn0 + j)*16 + lh;
            const float bv = bias[col];
#pragma unroll
            for (int r = 0; r < 8; ++r) {
                const int row = (tm0 + i)*16 + hi*8 + r;
                float v = acc[i][j][r] + bv;
                if (ACT) v = fmaxf(v, 0.f);
                if (OUT16) ((h16*)Out)[(size_t)row * ldo + col] = (h16)v;
                else       ((float*)Out)[(size_t)row * ldo + col] = v;
            }
        }
    }
}

// ---------------- small elementwise / staging kernels ----------------------
__global__ void f16pad_kernel(const float* __restrict__ x, h16* __restrict__ xh,
                              int M, int C, int Cp) {
    const int t = blockIdx.x * blockDim.x + threadIdx.x;
    if (t >= M * Cp) return;
    const int c = t % Cp, m = t / Cp;
    xh[t] = (c < C) ? (h16)x[(size_t)m * C + c] : (h16)0.f;
}

// W[O,I] f32 -> Wt[Ip,O] f16 (transpose + K-pad with zeros)
__global__ void wtrans_kernel(const float* __restrict__ W, h16* __restrict__ Wt,
                              int O, int I, int Ip) {
    const int t = blockIdx.x * blockDim.x + threadIdx.x;
    if (t >= Ip * O) return;
    const int n = t % O, k = t / O;
    Wt[t] = (k < I) ? (h16)W[(size_t)n * I + k] : (h16)0.f;
}

__global__ void zerofill_kernel(float* __restrict__ p, int n) {
    const int t = blockIdx.x * blockDim.x + threadIdx.x;
    if (t < n) p[t] = 0.f;
}

// pd[e,0:3] = pos[i]-pos[j], padded to 32 cols of f16 for the K=3 GEMM
__global__ void pd_kernel(const float* __restrict__ pos,
                          const int* __restrict__ idx,
                          h16* __restrict__ pdh, int N) {
    const int e = blockIdx.x * blockDim.x + threadIdx.x;
    if (e >= N * KNBR) return;
    const int n = e >> 4;
    const int j = idx[e];
    h16* o = pdh + (size_t)e * 32;
    o[0] = (h16)(pos[3*n]   - pos[3*j]);
    o[1] = (h16)(pos[3*n+1] - pos[3*j+1]);
    o[2] = (h16)(pos[3*n+2] - pos[3*j+2]);
#pragma unroll
    for (int c = 3; c < 32; ++c) o[c] = (h16)0.f;
}

// ain[e,c] = q[i,c] - k[j,c] + delta[e,c]   (f16 for the attention MLP)
__global__ void ain_kernel(const float* __restrict__ q,
                           const float* __restrict__ kk,
                           const float* __restrict__ delta,
                           const int* __restrict__ idx,
                           h16* __restrict__ ain, int N, int C) {
    const int t = blockIdx.x * blockDim.x + threadIdx.x;
    if (t >= N * KNBR * C) return;
    const int c = t % C;
    const int e = t / C;
    const int n = e >> 4;
    const int j = idx[e];
    ain[t] = (h16)(q[(size_t)n*C + c] - kk[(size_t)j*C + c] + delta[t]);
}

// out[n,c] = sum_k softmax_k(a[n,k,c]) * (v[idx[n,k],c] + delta[n,k,c])
__global__ void softagg_kernel(const float* __restrict__ a,
                               const float* __restrict__ delta,
                               const float* __restrict__ v,
                               const int* __restrict__ idx,
                               float* __restrict__ out, int N, int C) {
    const int t = blockIdx.x * blockDim.x + threadIdx.x;
    if (t >= N * C) return;
    const int c = t % C, n = t / C;
    const size_t base = (size_t)n * KNBR * C + c;
    float m = -3.0e38f;
#pragma unroll
    for (int k = 0; k < KNBR; ++k) m = fmaxf(m, a[base + (size_t)k*C]);
    float s = 0.f, acc = 0.f;
#pragma unroll
    for (int k = 0; k < KNBR; ++k) {
        const float e = __expf(a[base + (size_t)k*C] - m);
        const int j = idx[n*KNBR + k];
        s   += e;
        acc += e * (v[(size_t)j*C + c] + delta[base + (size_t)k*C]);
    }
    out[t] = acc / s;
}

// xo[no,c] = max(y[sel,c], max_k y[idx[sel,k],c]);  also subsample pos
__global__ void down_kernel(const float* __restrict__ y,
                            const int* __restrict__ idx,
                            const int* __restrict__ sel,
                            const float* __restrict__ pos,
                            float* __restrict__ xo, float* __restrict__ poso,
                            int Nout, int C) {
    const int t = blockIdx.x * blockDim.x + threadIdx.x;
    if (t >= Nout * C) return;
    const int c = t % C, no = t / C;
    const int n = sel[no];
    float m = y[(size_t)n*C + c];
#pragma unroll
    for (int k = 0; k < KNBR; ++k) {
        const int j = idx[n*KNBR + k];
        m = fmaxf(m, y[(size_t)j*C + c]);
    }
    xo[t] = m;
    if (c < 3) poso[no*3 + c] = pos[n*3 + c];
}

__global__ void mean_kernel(const float* __restrict__ x, float* __restrict__ g,
                            int N, int C) {
    const int c = blockIdx.x * blockDim.x + threadIdx.x;
    if (c >= C) return;
    float s = 0.f;
    for (int n = 0; n < N; ++n) s += x[(size_t)n*C + c];
    g[c] = s / (float)N;
}

__global__ void fc_kernel(const float* __restrict__ in,
                          const float* __restrict__ W,
                          const float* __restrict__ b,
                          float* __restrict__ out, int O, int I, int relu) {
    const int o = blockIdx.x * blockDim.x + threadIdx.x;
    if (o >= O) return;
    float s = b[o];
    for (int i = 0; i < I; ++i) s += in[i] * W[(size_t)o*I + i];
    out[o] = relu ? fmaxf(s, 0.f) : s;
}

__global__ void softmax40_kernel(const float* __restrict__ h,
                                 float* __restrict__ out, int n) {
    if (threadIdx.x == 0 && blockIdx.x == 0) {
        float m = h[0];
        for (int i = 1; i < n; ++i) m = fmaxf(m, h[i]);
        float s = 0.f;
        for (int i = 0; i < n; ++i) s += __expf(h[i] - m);
        for (int i = 0; i < n; ++i) out[i] = __expf(h[i] - m) / s;
    }
}

// ---------------------------------------------------------------------------
extern "C" void kernel_launch(void* const* d_in, const int* in_sizes, int n_in,
                              void* d_out, int out_size, void* d_ws,
                              size_t ws_size, hipStream_t stream) {
    (void)in_sizes; (void)n_in; (void)out_size; (void)ws_size;

    const float* x_in   = (const float*)d_in[0];
    const float* pos_in = (const float*)d_in[1];
    // input index map (dict flattening order)
    const int convBase[5] = { 2, 15, 28, 41, 54 };
    const int downBase[4] = { 13, 26, 39, 52 };
    const int headBase    = 65;
    const int selBase     = 71;

    // ------ workspace bump allocator (worst-case level-invariant sizes) ----
    char* ws = (char*)d_ws;
    size_t off = 0;
    auto alloc = [&](size_t bytes) -> void* {
        off = (off + 255) & ~(size_t)255;
        void* p = ws + off;
        off += bytes;
        return p;
    };
    const size_t NODE = 16384u * 32u;          // max N*C (invariant per level)
    const size_t EDGE = 16384u * 16u * 64u;    // max NK*2C (invariant)
    int*   idx   = (int*)  alloc(16384u * KNBR * 4);
    float* posA  = (float*)alloc(8192u * 3 * 4);
    float* posB  = (float*)alloc(8192u * 3 * 4);
    float* xA    = (float*)alloc(NODE * 4);
    float* xB    = (float*)alloc(NODE * 4);
    float* yT    = (float*)alloc(NODE * 4);
    float* qB    = (float*)alloc(NODE * 4);
    float* kB    = (float*)alloc(NODE * 4);
    float* vB    = (float*)alloc(NODE * 4);
    h16*   xh    = (h16*)  alloc(NODE * 2);
    h16*   Wt    = (h16*)  alloc(512u * 1024u * 2);
    float* zbias = (float*)alloc(1024u * 4);
    h16*   pdain = (h16*)  alloc((EDGE / 2) * 2);   // NK*C halves (== NK*32 at L1)
    h16*   hid   = (h16*)  alloc(EDGE * 2);         // NK*2C halves
    float* delta = (float*)alloc((EDGE / 2) * 4);   // NK*C f32
    float* af    = (float*)alloc((EDGE / 2) * 4);   // NK*C f32
    float* gV    = (float*)alloc(512 * 4);
    float* h1    = (float*)alloc(1024 * 4);
    float* h2    = (float*)alloc(512 * 4);
    float* h3    = (float*)alloc(64 * 4);

    auto cdiv = [](int a, int b) { return (a + b - 1) / b; };

    zerofill_kernel<<<cdiv(1024,256),256,0,stream>>>(zbias, 1024);

    auto launch_gemm = [&](const h16* A, int lda, const h16* Bm,
                           const float* bias, void* Out, int ldo,
                           int M, int N, int K, int act, bool f16out) {
        const int NTv    = (N % 64 == 0) ? 4 : 2;
        const int waves  = (M >> 5) * (N / (16 * NTv));
        const int blocks = cdiv(waves, 8);
        if (act && f16out) {
            if (NTv == 4) gemm_wmma<1,true ,4><<<blocks,256,0,stream>>>(A,lda,Bm,bias,Out,ldo,M,N,K);
            else          gemm_wmma<1,true ,2><<<blocks,256,0,stream>>>(A,lda,Bm,bias,Out,ldo,M,N,K);
        } else {
            if (NTv == 4) gemm_wmma<0,false,4><<<blocks,256,0,stream>>>(A,lda,Bm,bias,Out,ldo,M,N,K);
            else          gemm_wmma<0,false,2><<<blocks,256,0,stream>>>(A,lda,Bm,bias,Out,ldo,M,N,K);
        }
    };
    auto prep_w = [&](const void* W, int O, int I, int Ip) {
        wtrans_kernel<<<cdiv(Ip*O,256),256,0,stream>>>((const float*)W, Wt, O, I, Ip);
    };

    auto run_conv = [&](const float* x, const float* pos, int N,
                        int Cin, int Cout, int base, float* xout) {
        const int Cip = (Cin < 32) ? 32 : Cin;
        const int C2  = 2 * Cout;
        const int NK  = N * KNBR;
        const float* Wsrc = (const float*)d_in[base+0];
        const float* Wdst = (const float*)d_in[base+1];
        const float* Wlin = (const float*)d_in[base+2];
        const float* pW1  = (const float*)d_in[base+3];
        const float* pb1  = (const float*)d_in[base+4];
        const float* pW2  = (const float*)d_in[base+5];
        const float* pb2  = (const float*)d_in[base+6];
        const float* aW1  = (const float*)d_in[base+7];
        const float* ab1  = (const float*)d_in[base+8];
        const float* aW2  = (const float*)d_in[base+9];
        const float* ab2  = (const float*)d_in[base+10];

        // stage x -> f16 (K-padded)
        f16pad_kernel<<<cdiv(N*Cip,256),256,0,stream>>>(x, xh, N, Cin, Cip);
        // q, k, v projections (WMMA)
        prep_w(Wsrc, Cout, Cin, Cip);
        launch_gemm(xh, Cip, Wt, zbias, qB, Cout, N, Cout, Cip, 0, false);
        prep_w(Wdst, Cout, Cin, Cip);
        launch_gemm(xh, Cip, Wt, zbias, kB, Cout, N, Cout, Cip, 0, false);
        prep_w(Wlin, Cout, Cin, Cip);
        launch_gemm(xh, Cip, Wt, zbias, vB, Cout, N, Cout, Cip, 0, false);
        // pos_nn MLP over edges (WMMA x2, fused bias+ReLU, f16 hidden)
        pd_kernel<<<cdiv(NK,256),256,0,stream>>>(pos, idx, pdain, N);
        prep_w(pW1, C2, 3, 32);
        launch_gemm(pdain, 32, Wt, pb1, hid, C2, NK, C2, 32, 1, true);
        prep_w(pW2, Cout, C2, C2);
        launch_gemm(hid, C2, Wt, pb2, delta, Cout, NK, Cout, C2, 0, false);
        // attn_nn MLP over edges (WMMA x2)
        ain_kernel<<<cdiv(NK*Cout,256),256,0,stream>>>(qB, kB, delta, idx, pdain, N, Cout);
        prep_w(aW1, C2, Cout, Cout);
        launch_gemm(pdain, Cout, Wt, ab1, hid, C2, NK, C2, Cout, 1, true);
        prep_w(aW2, Cout, C2, C2);
        launch_gemm(hid, C2, Wt, ab2, af, Cout, NK, Cout, C2, 0, false);
        // softmax over neighbors + weighted aggregation
        softagg_kernel<<<cdiv(N*Cout,256),256,0,stream>>>(af, delta, vB, idx, xout, N, Cout);
    };

    auto run_down = [&](const float* x, const float* pos, const int* sel,
                        int N, int C, int base, float* xout, float* posout) {
        f16pad_kernel<<<cdiv(N*C,256),256,0,stream>>>(x, xh, N, C, C);
        prep_w(d_in[base+0], C, C, C);
        launch_gemm(xh, C, Wt, (const float*)d_in[base+1], yT, C, N, C, C, 0, false);
        down_kernel<<<cdiv((N/2)*C,256),256,0,stream>>>(yT, idx, sel, pos, xout, posout, N/2, C);
    };

    // ------------------------------- pipeline -----------------------------
    knn_kernel<<<cdiv(16384,256),256,0,stream>>>(pos_in, 16384, idx);
    run_conv(x_in, pos_in, 16384, 3, 32, convBase[0], xA);
    run_down(xA, pos_in, (const int*)d_in[selBase+0], 16384, 32, downBase[0], xB, posA);

    knn_kernel<<<cdiv(8192,256),256,0,stream>>>(posA, 8192, idx);
    run_conv(xB, posA, 8192, 32, 64, convBase[1], xA);
    run_down(xA, posA, (const int*)d_in[selBase+1], 8192, 64, downBase[1], xB, posB);

    knn_kernel<<<cdiv(4096,256),256,0,stream>>>(posB, 4096, idx);
    run_conv(xB, posB, 4096, 64, 128, convBase[2], xA);
    run_down(xA, posB, (const int*)d_in[selBase+2], 4096, 128, downBase[2], xB, posA);

    knn_kernel<<<cdiv(2048,256),256,0,stream>>>(posA, 2048, idx);
    run_conv(xB, posA, 2048, 128, 256, convBase[3], xA);
    run_down(xA, posA, (const int*)d_in[selBase+3], 2048, 256, downBase[3], xB, posB);

    knn_kernel<<<cdiv(1024,256),256,0,stream>>>(posB, 1024, idx);
    run_conv(xB, posB, 1024, 256, 512, convBase[4], xA);   // [1024, 512]

    // --------------------------------- head -------------------------------
    const float* W1 = (const float*)d_in[headBase+0];
    const float* b1 = (const float*)d_in[headBase+1];
    const float* W2 = (const float*)d_in[headBase+2];
    const float* b2 = (const float*)d_in[headBase+3];
    const float* W3 = (const float*)d_in[headBase+4];
    const float* b3 = (const float*)d_in[headBase+5];
    mean_kernel<<<cdiv(512,256),256,0,stream>>>(xA, gV, 1024, 512);
    fc_kernel<<<cdiv(1024,256),256,0,stream>>>(gV, W1, b1, h1, 1024, 512, 1);
    fc_kernel<<<cdiv(512,256),256,0,stream>>>(h1, W2, b2, h2, 512, 1024, 0);
    fc_kernel<<<1,64,0,stream>>>(h2, W3, b3, h3, 40, 512, 1);
    softmax40_kernel<<<1,64,0,stream>>>(h3, (float*)d_out, 40);
}
